// Mamba2Block_4363686773536
// MI455X (gfx1250) — matrix operand order repaired and checked by reading, not compiled, not run on hardware
//
#include <hip/hip_runtime.h>
#include <cstddef>

// ---------------- constants (match reference) ----------------
#define D_MODEL   1024
#define D_STATE   128
#define HEAD_DIM  64
#define D_INNER   2048
#define N_HEADS   32
#define D_PROJ    4384        // 2*D_INNER + 2*D_STATE + N_HEADS
#define NBATCH    2
#define LSEQ      2048
#define NROWS     (NBATCH*LSEQ)   // 4096
#define EPS_RMS   1e-6f

// ---------------- WMMA vector types ----------------
typedef __attribute__((ext_vector_type(16))) __bf16 v16bf;
typedef __attribute__((ext_vector_type(8)))  __bf16 v8bf;
typedef __attribute__((ext_vector_type(8)))  float  v8f;

// =================================================================
// K0: RMSNorm -> bf16   (one block per row, 256 threads)
// =================================================================
__global__ __launch_bounds__(256) void k_rmsnorm(const float* __restrict__ x,
                                                 const float* __restrict__ w,
                                                 __bf16* __restrict__ xn) {
  const int row = blockIdx.x;
  const int t = threadIdx.x;
  const float* xr = x + (size_t)row * D_MODEL;
  float v[4];
  float s = 0.f;
#pragma unroll
  for (int i = 0; i < 4; ++i) { v[i] = xr[t + 256 * i]; s += v[i] * v[i]; }
#pragma unroll
  for (int m = 1; m < 32; m <<= 1) s += __shfl_xor(s, m, 32);
  __shared__ float red[8];
  if ((t & 31) == 0) red[t >> 5] = s;
  __syncthreads();
  float tot = 0.f;
#pragma unroll
  for (int i = 0; i < 8; ++i) tot += red[i];
  const float inv = rsqrtf(tot * (1.0f / D_MODEL) + EPS_RMS);
  __bf16* outr = xn + (size_t)row * D_MODEL;
#pragma unroll
  for (int i = 0; i < 4; ++i)
    outr[t + 256 * i] = (__bf16)(v[i] * inv * w[t + 256 * i]);
}

// =================================================================
// K1: fp32 -> bf16 elementwise convert
// =================================================================
__global__ void k_f2bf(const float* __restrict__ in, __bf16* __restrict__ out, int n) {
  int i = blockIdx.x * 256 + threadIdx.x;
  if (i < n) out[i] = (__bf16)in[i];
}

// =================================================================
// K2/K6: bf16 WMMA GEMM:  out[M][N] = A[M][K] * W[N][K]^T (+ resid)
// One wave -> 32(M) x 32(N) tile (4 accumulators, 2 loads per WMMA).
// Block = 8 waves = 256 M rows. grid.x = M/256, grid.y = N/32.
// =================================================================
__global__ __launch_bounds__(256) void k_gemm_bf16(
    const __bf16* __restrict__ A,     // [M][K] row-major
    const __bf16* __restrict__ W,     // [N][K] row-major (weight rows)
    const float*  __restrict__ resid, // nullptr or [M][N]
    float* __restrict__ out,          // [M][N]
    int K, int Nout) {
  const int lane = threadIdx.x & 31;
  const int wave = threadIdx.x >> 5;
  const int hf   = lane >> 4;     // lane half (0/1)
  const int lr   = lane & 15;
  const int mbase = (blockIdx.x * 8 + wave) * 32;
  const int nbase = blockIdx.y * 32;

  const __bf16* Arow0 = A + (size_t)(mbase + lr) * K;
  const __bf16* Arow1 = A + (size_t)(mbase + 16 + lr) * K;
  const __bf16* Wc0   = W + (size_t)(nbase + lr) * K;
  const __bf16* Wc1   = W + (size_t)(nbase + 16 + lr) * K;

  v8f acc00 = {0.f,0.f,0.f,0.f,0.f,0.f,0.f,0.f};
  v8f acc01 = {0.f,0.f,0.f,0.f,0.f,0.f,0.f,0.f};
  v8f acc10 = {0.f,0.f,0.f,0.f,0.f,0.f,0.f,0.f};
  v8f acc11 = {0.f,0.f,0.f,0.f,0.f,0.f,0.f,0.f};

  for (int k = 0; k < K; k += 32) {
    // A fragment (16x32 bf16): lane<16 holds K[k..k+7],K[k+16..k+23] of row M=lr
    //                          lane>=16 holds K[k+8..k+15],K[k+24..k+31]
    v8bf a0lo = *(const v8bf*)(Arow0 + k + hf * 8);
    v8bf a0hi = *(const v8bf*)(Arow0 + k + 16 + hf * 8);
    v8bf a1lo = *(const v8bf*)(Arow1 + k + hf * 8);
    v8bf a1hi = *(const v8bf*)(Arow1 + k + 16 + hf * 8);
    // B fragment (32x16 bf16): lane<16 holds K[k..k+7],K[k+16..k+23] of col N=lr,
    //                          lane>=16 holds K[k+8..k+15],K[k+24..k+31]
    v8bf b0l = *(const v8bf*)(Wc0 + k + hf * 8);
    v8bf b0h = *(const v8bf*)(Wc0 + k + 16 + hf * 8);
    v8bf b1l = *(const v8bf*)(Wc1 + k + hf * 8);
    v8bf b1h = *(const v8bf*)(Wc1 + k + 16 + hf * 8);

    v16bf a0, a1, b0, b1;
#pragma unroll
    for (int i = 0; i < 8; ++i) {
      a0[i] = a0lo[i]; a0[8 + i] = a0hi[i];
      a1[i] = a1lo[i]; a1[8 + i] = a1hi[i];
      b0[i] = b0l[i];  b0[8 + i] = b0h[i];
      b1[i] = b1l[i];  b1[8 + i] = b1h[i];
    }

    acc00 = __builtin_amdgcn_wmma_f32_16x16x32_bf16(false, a0, false, b0,
                                                    (short)0, acc00, false, false);
    acc01 = __builtin_amdgcn_wmma_f32_16x16x32_bf16(false, a0, false, b1,
                                                    (short)0, acc01, false, false);
    acc10 = __builtin_amdgcn_wmma_f32_16x16x32_bf16(false, a1, false, b0,
                                                    (short)0, acc10, false, false);
    acc11 = __builtin_amdgcn_wmma_f32_16x16x32_bf16(false, a1, false, b1,
                                                    (short)0, acc11, false, false);
  }

  // D layout: VGPR r -> M = hf*8 + r, N = lr (per 16-lane half)
  const int col0 = nbase + lr;
  const int col1 = nbase + 16 + lr;
#pragma unroll
  for (int r = 0; r < 8; ++r) {
    const int row0 = mbase + hf * 8 + r;
    const int row1 = mbase + 16 + hf * 8 + r;
    const size_t o00 = (size_t)row0 * Nout + col0;
    const size_t o01 = (size_t)row0 * Nout + col1;
    const size_t o10 = (size_t)row1 * Nout + col0;
    const size_t o11 = (size_t)row1 * Nout + col1;
    float r00 = acc00[r], r01 = acc01[r], r10 = acc10[r], r11 = acc11[r];
    if (resid) {
      r00 += resid[o00]; r01 += resid[o01];
      r10 += resid[o10]; r11 += resid[o11];
    }
    out[o00] = r00; out[o01] = r01;
    out[o10] = r10; out[o11] = r11;
  }
}

// =================================================================
// K3: causal depthwise conv (K=4) + bias + SiLU on x_ssm slice of proj
// =================================================================
__global__ void k_conv(const float* __restrict__ proj, const float* __restrict__ cw,
                       const float* __restrict__ cb, float* __restrict__ xconv) {
  const int i = blockIdx.x * 256 + threadIdx.x;     // over NROWS*D_INNER
  if (i >= NROWS * D_INNER) return;
  const int c = i & (D_INNER - 1);
  const int row = i >> 11;                          // D_INNER = 2^11
  const int l = row & (LSEQ - 1);
  float acc = cb[c];
#pragma unroll
  for (int k = 0; k < 4; ++k) {
    const int ls = l - 3 + k;
    if (ls >= 0)
      acc += proj[(size_t)(row - 3 + k) * D_PROJ + 2048 + c] * cw[c * 4 + k];
  }
  const float sg = 1.f / (1.f + __expf(-acc));
  xconv[i] = acc * sg;
}

// =================================================================
// K3b: dt = softplus(dt_raw + bias); dA = exp(-exp(A_log)*dt)
// =================================================================
__global__ void k_dt(const float* __restrict__ proj, const float* __restrict__ dt_bias,
                     const float* __restrict__ A_log,
                     float* __restrict__ dt, float* __restrict__ dA) {
  const int i = blockIdx.x * 256 + threadIdx.x;     // over NROWS*N_HEADS
  if (i >= NROWS * N_HEADS) return;
  const int h = i & (N_HEADS - 1);
  const int row = i >> 5;
  const float s = proj[(size_t)row * D_PROJ + 4352 + h] + dt_bias[h];
  const float sp = (s > 20.f) ? s : log1pf(__expf(s));
  dt[i] = sp;
  dA[i] = __expf(-__expf(A_log[h]) * sp);
}

// =================================================================
// K4: sequential SSM scan. One 512-thread WG per (batch, head).
// Thread t owns p = t>>3 (0..63), n-chunk = (t&7)*16 -> 16 state regs.
// B/C rows staged in double-buffered LDS; one barrier per step.
// =================================================================
__global__ __launch_bounds__(512) void k_scan(
    const float* __restrict__ proj, const float* __restrict__ xconv,
    const float* __restrict__ dt, const float* __restrict__ dA,
    float* __restrict__ yscan) {
  __shared__ float bc[2][256];
  const int bh = blockIdx.x;
  const int b = bh >> 5, h = bh & (N_HEADS - 1);
  const int t = threadIdx.x;
  const int p = t >> 3;
  const int cg = t & 7;
  const int n0 = cg * 16;

  float hs[16];
#pragma unroll
  for (int j = 0; j < 16; ++j) hs[j] = 0.f;

  for (int l = 0; l < LSEQ; ++l) {
    const size_t row = (size_t)(b * LSEQ + l);
    const float* pr = proj + row * D_PROJ;
    float* buf = bc[l & 1];
    if (t < 128)      buf[t] = pr[4096 + t];          // B (group 0)
    else if (t < 256) buf[t] = pr[4224 + (t - 128)];  // C (group 0)
    __syncthreads();

    const float dAl = dA[row * N_HEADS + h];
    const float dtl = dt[row * N_HEADS + h];
    const float xh  = xconv[row * D_INNER + h * HEAD_DIM + p];
    const float dtx = dtl * xh;

    const float4* Bv = (const float4*)(buf + n0);
    const float4* Cv = (const float4*)(buf + 128 + n0);
    float y = 0.f;
#pragma unroll
    for (int q = 0; q < 4; ++q) {
      const float4 Bq = Bv[q];
      const float4 Cq = Cv[q];
      hs[4*q+0] = hs[4*q+0] * dAl + dtx * Bq.x;  y += hs[4*q+0] * Cq.x;
      hs[4*q+1] = hs[4*q+1] * dAl + dtx * Bq.y;  y += hs[4*q+1] * Cq.y;
      hs[4*q+2] = hs[4*q+2] * dAl + dtx * Bq.z;  y += hs[4*q+2] * Cq.z;
      hs[4*q+3] = hs[4*q+3] * dAl + dtx * Bq.w;  y += hs[4*q+3] * Cq.w;
    }
    y += __shfl_xor(y, 1, 32);
    y += __shfl_xor(y, 2, 32);
    y += __shfl_xor(y, 4, 32);
    if (cg == 0) yscan[row * D_INNER + h * HEAD_DIM + p] = y;
    // no trailing barrier: double-buffered LDS makes next step's writes safe
  }
}

// =================================================================
// K5: y = (yscan + D[h]*xh) * silu(z)  -> bf16 for the output GEMM
// =================================================================
__global__ void k_gate(const float* __restrict__ proj, const float* __restrict__ yscan,
                       const float* __restrict__ xconv, const float* __restrict__ Dp,
                       __bf16* __restrict__ ybf) {
  const int i = blockIdx.x * 256 + threadIdx.x;     // over NROWS*D_INNER
  if (i >= NROWS * D_INNER) return;
  const int c = i & (D_INNER - 1);
  const int row = i >> 11;
  const int h = c >> 6;
  const float z = proj[(size_t)row * D_PROJ + c];
  const float y = yscan[i] + Dp[h] * xconv[i];
  const float sg = z / (1.f + __expf(-z));
  ybf[i] = (__bf16)(y * sg);
}

// =================================================================
// launch
// =================================================================
extern "C" void kernel_launch(void* const* d_in, const int* in_sizes, int n_in,
                              void* d_out, int out_size, void* d_ws, size_t ws_size,
                              hipStream_t stream) {
  const float* x          = (const float*)d_in[0];
  const float* norm_w     = (const float*)d_in[1];
  const float* in_proj_w  = (const float*)d_in[2];
  const float* conv_w     = (const float*)d_in[3];
  const float* conv_b     = (const float*)d_in[4];
  const float* A_log      = (const float*)d_in[5];
  const float* Dvec       = (const float*)d_in[6];
  const float* dt_bias    = (const float*)d_in[7];
  const float* out_proj_w = (const float*)d_in[8];
  float* out = (float*)d_out;

  char* ws = (char*)d_ws;
  size_t off = 0;
  auto alloc = [&](size_t bytes) -> void* {
    void* p = ws + off;
    off = (off + bytes + 255) & ~(size_t)255;
    return p;
  };
  __bf16* Wa    = (__bf16*)alloc((size_t)D_PROJ * D_MODEL * 2);
  __bf16* Wo    = (__bf16*)alloc((size_t)D_MODEL * D_INNER * 2);
  __bf16* xn    = (__bf16*)alloc((size_t)NROWS * D_MODEL * 2);
  float*  proj  = (float*)alloc((size_t)NROWS * D_PROJ * 4);
  float*  xconv = (float*)alloc((size_t)NROWS * D_INNER * 4);
  float*  dt    = (float*)alloc((size_t)NROWS * N_HEADS * 4);
  float*  dA    = (float*)alloc((size_t)NROWS * N_HEADS * 4);
  float*  yscan = (float*)alloc((size_t)NROWS * D_INNER * 4);
  __bf16* ybf   = (__bf16*)alloc((size_t)NROWS * D_INNER * 2);

  k_rmsnorm<<<NROWS, 256, 0, stream>>>(x, norm_w, xn);
  k_f2bf<<<(D_PROJ * D_MODEL + 255) / 256, 256, 0, stream>>>(in_proj_w, Wa,
                                                             D_PROJ * D_MODEL);
  k_f2bf<<<(D_MODEL * D_INNER + 255) / 256, 256, 0, stream>>>(out_proj_w, Wo,
                                                              D_MODEL * D_INNER);
  {
    dim3 g(NROWS / 256, D_PROJ / 32);   // 16 x 137
    k_gemm_bf16<<<g, 256, 0, stream>>>(xn, Wa, nullptr, proj, D_MODEL, D_PROJ);
  }
  k_conv<<<(NROWS * D_INNER + 255) / 256, 256, 0, stream>>>(proj, conv_w, conv_b,
                                                            xconv);
  k_dt<<<(NROWS * N_HEADS + 255) / 256, 256, 0, stream>>>(proj, dt_bias, A_log,
                                                          dt, dA);
  k_scan<<<NBATCH * N_HEADS, 512, 0, stream>>>(proj, xconv, dt, dA, yscan);
  k_gate<<<(NROWS * D_INNER + 255) / 256, 256, 0, stream>>>(proj, yscan, xconv,
                                                            Dvec, ybf);
  {
    dim3 g(NROWS / 256, D_MODEL / 32);  // 16 x 32
    k_gemm_bf16<<<g, 256, 0, stream>>>(ybf, Wo, x, out, D_INNER, D_MODEL);
  }
}
